// TemporalAttention_43473658970666
// MI455X (gfx1250) — matrix operand order repaired and checked by reading, not compiled, not run on hardware
//
#include <hip/hip_runtime.h>
#include <hip/hip_bf16.h>

// ---------------------------------------------------------------------------
// Fused multi-head self-attention block (proj -> attention -> proj -> LN)
// for MI455X / gfx1250: bf16 WMMA (V_WMMA_F32_16X16X32_BF16) everywhere,
// GLOBAL_LOAD_ASYNC_TO_LDS_B128 staging for shared A-operands,
// big-LDS (192KB/WG) exact softmax.
// ---------------------------------------------------------------------------

typedef __bf16 bf16_t;
typedef __attribute__((ext_vector_type(16))) __bf16 v16bf;
typedef __attribute__((ext_vector_type(8)))  __bf16 v8bf;
typedef __attribute__((ext_vector_type(8)))  float  v8f;

#define D_MODEL 512
#define NHEADS  8
#define HDIM    64
#define SEQ     2048
#define BATCH   4
#define ROWS    (BATCH * SEQ)      // 8192 token rows

// ---- WMMA fragment helpers (wave32, contiguous k = 16h + s order, same in A and B) ----------

// A-matrix 16x32 bf16. Lane l<16: row M=l, elems 0..15 = K(kb..kb+15).
// Lane l>=16: row M=l-16, elems 0..15 = K(kb+16..kb+31).
__device__ __forceinline__ v16bf load_a16x32(const bf16_t* src, int ld, int kb) {
    int lane = threadIdx.x & 31;
    int hb   = lane >> 4;
    int row  = lane & 15;
    const bf16_t* p0 = src + row * ld + kb + hb * 16;
    v8bf c0 = *(const v8bf*)(p0);
    v8bf c1 = *(const v8bf*)(p0 + 8);
    v16bf a;
#pragma unroll
    for (int i = 0; i < 8; ++i) { a[i] = c0[i]; a[i + 8] = c1[i]; }
    return a;
}

// B-matrix 32x16 bf16, sourced from B-transposed row-major storage [N][K]:
// lane l<16: column N=l, K=kb..kb+15 ; lane l>=16: column N=l-16, K=kb+16..kb+31.
// => one contiguous 32-byte load per lane.
__device__ __forceinline__ v16bf load_bT32x16(const bf16_t* bt, int ld, int kb) {
    int lane = threadIdx.x & 31;
    int hb   = lane >> 4;
    int col  = lane & 15;
    return *(const v16bf*)(bt + col * ld + kb + hb * 16);
}

__device__ __forceinline__ v8f wmma_bf16(v16bf a, v16bf b, v8f c) {
    return __builtin_amdgcn_wmma_f32_16x16x32_bf16(
        /*neg_a=*/false, a, /*neg_b=*/false, b,
        /*c_mod=*/(short)0, c, /*reuse_a=*/false, /*reuse_b=*/false);
}

// Async-copy `bytes16` 16-byte chunks from global to LDS using the gfx1250
// ASYNCcnt path (GLOBAL_LOAD_ASYNC_TO_LDS_B128). Each thread copies one chunk
// per iteration. Caller must follow with async_wait() + __syncthreads().
__device__ __forceinline__ void async_stage_16b(const void* gsrc, void* lds_dst,
                                                int chunk_idx) {
    unsigned int  la = (unsigned int)(unsigned long long)lds_dst + chunk_idx * 16u;
    unsigned long long ga = (unsigned long long)gsrc + (unsigned long long)chunk_idx * 16ull;
    asm volatile("global_load_async_to_lds_b128 %0, %1, off"
                 :: "v"(la), "v"(ga) : "memory");
}
__device__ __forceinline__ void async_wait0() {
    asm volatile("s_wait_asynccnt 0x0" ::: "memory");
}

// ---------------------------------------------------------------------------
// Kernel 1: convert x and the four weight matrices to bf16 scratch.
// ---------------------------------------------------------------------------
__global__ void prep_kernel(const float* __restrict__ x,
                            const float* __restrict__ wq, const float* __restrict__ wk,
                            const float* __restrict__ wv, const float* __restrict__ wo,
                            bf16_t* __restrict__ xb,
                            bf16_t* __restrict__ wqb, bf16_t* __restrict__ wkb,
                            bf16_t* __restrict__ wvb, bf16_t* __restrict__ wob) {
    int i = blockIdx.x * blockDim.x + threadIdx.x;
    int stride = gridDim.x * blockDim.x;
    const int NX = ROWS * D_MODEL;
    const int NW = D_MODEL * D_MODEL;
    for (int j = i; j < NX; j += stride) xb[j] = (bf16_t)x[j];
    for (int j = i; j < NW; j += stride) {
        wqb[j] = (bf16_t)wq[j];
        wkb[j] = (bf16_t)wk[j];
        wvb[j] = (bf16_t)wv[j];
        wob[j] = (bf16_t)wo[j];
    }
}

// ---------------------------------------------------------------------------
// Kernel 2: QKV projection. One block = one 16-row m-stripe of x, staged into
// LDS once via async-to-LDS; its 8 waves cover all 96 output tiles
// (3 outputs x 32 n-tiles), 3 groups of 4 tiles each with A-fragment reuse.
// Q,K stored [B,H,S,HD]; V stored transposed [B,H,HD,S] for the PV GEMM.
// ---------------------------------------------------------------------------
__global__ void qkv_gemm_kernel(const bf16_t* __restrict__ xb,
                                const bf16_t* __restrict__ wqb,
                                const bf16_t* __restrict__ wkb,
                                const bf16_t* __restrict__ wvb,
                                const float* __restrict__ bq,
                                const float* __restrict__ bk,
                                const float* __restrict__ bv,
                                bf16_t* __restrict__ qb,
                                bf16_t* __restrict__ kb,
                                bf16_t* __restrict__ vt) {
    __shared__ bf16_t Atile[16 * D_MODEL];   // 16KB: the block's A-stripe

    int mt = blockIdx.x;                     // 0..511 m-tile
    const bf16_t* src = xb + (size_t)(mt * 16) * D_MODEL;

    // Stage 16x512 bf16 (16KB = 1024 x 16B chunks) into LDS asynchronously.
#pragma unroll
    for (int it = 0; it < 4; ++it)
        async_stage_16b(src, Atile, it * 256 + (int)threadIdx.x);
    async_wait0();
    __syncthreads();

    int wave = threadIdx.x >> 5;
    int lane = threadIdx.x & 31;
    int hb   = lane >> 4;
    int col  = lane & 15;

#pragma unroll
    for (int g = 0; g < 3; ++g) {
        int t0   = wave * 12 + g * 4;        // first of 4 consecutive tiles
        int osel = t0 >> 5;                  // 0=Q 1=K 2=V (group never spans)
        int ntb  = t0 & 31;                  // base n-tile within that output
        const bf16_t* W    = (osel == 0) ? wqb : ((osel == 1) ? wkb : wvb);
        const float*  bias = (osel == 0) ? bq  : ((osel == 1) ? bk  : bv);

        v8f acc[4] = {};
#pragma unroll 4
        for (int kk = 0; kk < D_MODEL; kk += 32) {
            v16bf a = load_a16x32(Atile, D_MODEL, kk);   // from LDS, reused x4
#pragma unroll
            for (int t = 0; t < 4; ++t) {
                v16bf b = load_bT32x16(W + (size_t)((ntb + t) * 16) * D_MODEL,
                                       D_MODEL, kk);
                acc[t] = wmma_bf16(a, b, acc[t]);
            }
        }

#pragma unroll
        for (int t = 0; t < 4; ++t) {
            int n  = (ntb + t) * 16 + col;
            float bi = bias[n];
            int h  = n >> 6;
            int dd = n & (HDIM - 1);
#pragma unroll
            for (int r = 0; r < 8; ++r) {
                int m   = mt * 16 + r + hb * 8;
                int bat = m >> 11;
                int s   = m & (SEQ - 1);
                float val = acc[t][r] + bi;
                if (osel == 2) {
                    vt[((size_t)(bat * NHEADS + h) * HDIM + dd) * SEQ + s] = (bf16_t)val;
                } else {
                    bf16_t* dst = (osel == 0) ? qb : kb;
                    dst[((size_t)(bat * NHEADS + h) * SEQ + s) * HDIM + dd] = (bf16_t)val;
                }
            }
        }
    }
}

// ---------------------------------------------------------------------------
// Kernel 3: attention. One workgroup (4 waves, 128 threads) owns one
// (batch, head, 16-query) stripe. Full 16x2048 f32 score stripe (128KB) and
// 16x2048 bf16 probability stripe (64KB) live in dynamic LDS — exploits
// CDNA5's 320KB/WGP LDS. Exact two-pass softmax; context accumulated with
// unnormalized exp, divided by row-sum at the end.
// ---------------------------------------------------------------------------
__global__ void attn_kernel(const bf16_t* __restrict__ qb,
                            const bf16_t* __restrict__ kb,
                            const bf16_t* __restrict__ vt,
                            bf16_t* __restrict__ cb) {
    extern __shared__ __align__(16) char smem[];
    float*  sc = (float*)smem;                         // [16][SEQ] f32 scores
    bf16_t* P  = (bf16_t*)(smem + 16 * SEQ * 4);       // [16][SEQ] bf16 exp

    __shared__ float red[16][8];
    __shared__ float rowmax[16];
    __shared__ float rowinv[16];

    int bh = blockIdx.x / (SEQ / 16);
    int qt = blockIdx.x % (SEQ / 16);
    int bat = bh / NHEADS;
    int h   = bh % NHEADS;

    const bf16_t* qh = qb + (size_t)bh * SEQ * HDIM;   // [S][HD]
    const bf16_t* kh = kb + (size_t)bh * SEQ * HDIM;   // [S][HD]  (= B^T of K^T)
    const bf16_t* vh = vt + (size_t)bh * HDIM * SEQ;   // [HD][S]  (= B^T of V)

    int wave = threadIdx.x >> 5;
    int lane = threadIdx.x & 31;
    int hb   = lane >> 4;
    int lcol = lane & 15;

    // Q tile fragments reused across all 128 key tiles.
    const bf16_t* qtile = qh + (size_t)(qt * 16) * HDIM;
    v16bf aq0 = load_a16x32(qtile, HDIM, 0);
    v16bf aq1 = load_a16x32(qtile, HDIM, 32);

    // ---- Phase 1: S = (Q K^T) / sqrt(HD) into LDS -------------------------
    for (int jt = wave; jt < SEQ / 16; jt += 4) {
        const bf16_t* ktile = kh + (size_t)(jt * 16) * HDIM;
        v16bf b0 = load_bT32x16(ktile, HDIM, 0);
        v16bf b1 = load_bT32x16(ktile, HDIM, 32);
        v8f acc = {};
        acc = wmma_bf16(aq0, b0, acc);
        acc = wmma_bf16(aq1, b1, acc);
#pragma unroll
        for (int r = 0; r < 8; ++r) {
            int M = r + hb * 8;
            sc[M * SEQ + jt * 16 + lcol] = acc[r] * 0.125f;  // 1/sqrt(64)
        }
    }
    __syncthreads();

    // ---- Phase 2: softmax (rowmax, unnormalized exp -> bf16, rowsum) ------
    {
        int r  = threadIdx.x >> 3;   // 0..15 : query row
        int sl = threadIdx.x & 7;    // 0..7  : slice of 256 keys
        const int CH = SEQ / 8;
        float mx = -3.4e38f;
        for (int j = sl * CH; j < (sl + 1) * CH; ++j) mx = fmaxf(mx, sc[r * SEQ + j]);
        red[r][sl] = mx;
        __syncthreads();
        if (sl == 0) {
            float m2 = red[r][0];
            for (int i = 1; i < 8; ++i) m2 = fmaxf(m2, red[r][i]);
            rowmax[r] = m2;
        }
        __syncthreads();
        float rm = rowmax[r];
        float sum = 0.f;
        for (int j = sl * CH; j < (sl + 1) * CH; ++j) {
            float e = __expf(sc[r * SEQ + j] - rm);
            P[r * SEQ + j] = (bf16_t)e;
            sum += e;
        }
        red[r][sl] = sum;
        __syncthreads();
        if (sl == 0) {
            float s2 = 0.f;
            for (int i = 0; i < 8; ++i) s2 += red[r][i];
            rowinv[r] = 1.f / s2;
        }
        __syncthreads();
    }

    // ---- Phase 3: context = P @ V ; each wave owns one 16-wide d slice ----
    int db = wave * 16;
    v8f acc = {};
    for (int jb = 0; jb < SEQ; jb += 32) {
        v16bf a = load_a16x32(P, SEQ, jb);                      // A from LDS
        v16bf b = load_bT32x16(vh + (size_t)db * SEQ, SEQ, jb); // B^T = V^T
        acc = wmma_bf16(a, b, acc);
    }
#pragma unroll
    for (int r = 0; r < 8; ++r) {
        int M = r + hb * 8;
        float val = acc[r] * rowinv[M];
        int s = qt * 16 + M;
        int n = h * HDIM + db + lcol;
        cb[((size_t)(bat * SEQ + s)) * D_MODEL + n] = (bf16_t)val;
    }
}

// ---------------------------------------------------------------------------
// Kernel 4: output projection + bias + residual + LayerNorm.
// One block (8 waves) = 16 token rows x full 512 cols; context A-stripe staged
// into LDS via async-to-LDS; each wave computes 4 n-tiles with A reuse.
// ---------------------------------------------------------------------------
__global__ void out_ln_kernel(const bf16_t* __restrict__ cb,
                              const bf16_t* __restrict__ wob,
                              const float* __restrict__ bo,
                              const float* __restrict__ x,
                              const float* __restrict__ gamma,
                              const float* __restrict__ beta,
                              float* __restrict__ out) {
    __shared__ bf16_t Atile[16 * D_MODEL];   // 16KB context A-stripe
    __shared__ float y[16][D_MODEL];         // 32KB staging for LN
    __shared__ float red2[16][16][2];
    __shared__ float mu[16], rs[16];

    int mt   = blockIdx.x;
    int wave = threadIdx.x >> 5;
    int lane = threadIdx.x & 31;
    int hb   = lane >> 4;
    int col  = lane & 15;

    const bf16_t* src = cb + (size_t)(mt * 16) * D_MODEL;
#pragma unroll
    for (int it = 0; it < 4; ++it)
        async_stage_16b(src, Atile, it * 256 + (int)threadIdx.x);
    async_wait0();
    __syncthreads();

    v8f acc[4] = {};
#pragma unroll 4
    for (int kk = 0; kk < D_MODEL; kk += 32) {
        v16bf a = load_a16x32(Atile, D_MODEL, kk);
#pragma unroll
        for (int t = 0; t < 4; ++t) {
            int nt = wave * 4 + t;
            v16bf b = load_bT32x16(wob + (size_t)(nt * 16) * D_MODEL, D_MODEL, kk);
            acc[t] = wmma_bf16(a, b, acc[t]);   // A reused across 4 B tiles
        }
    }
#pragma unroll
    for (int t = 0; t < 4; ++t) {
        int n = (wave * 4 + t) * 16 + col;
        float bi = bo[n];
#pragma unroll
        for (int r = 0; r < 8; ++r) {
            int M = r + hb * 8;
            int m = mt * 16 + M;
            y[M][n] = acc[t][r] + bi + x[(size_t)m * D_MODEL + n];  // +residual
        }
    }
    __syncthreads();

    int r  = threadIdx.x >> 4;   // 0..15 row
    int sl = threadIdx.x & 15;   // 0..15 slice of 32
    float s = 0.f, s2 = 0.f;
    for (int j = sl * 32; j < sl * 32 + 32; ++j) { float v = y[r][j]; s += v; s2 += v * v; }
    red2[r][sl][0] = s; red2[r][sl][1] = s2;
    __syncthreads();
    if (sl == 0) {
        float a = 0.f, b = 0.f;
        for (int i = 0; i < 16; ++i) { a += red2[r][i][0]; b += red2[r][i][1]; }
        float mean = a * (1.f / D_MODEL);
        float var  = b * (1.f / D_MODEL) - mean * mean;
        mu[r] = mean;
        rs[r] = rsqrtf(var + 1e-5f);
    }
    __syncthreads();
    float mean = mu[r], rstd = rs[r];
    for (int j = sl * 32; j < sl * 32 + 32; ++j) {
        out[(size_t)(mt * 16 + r) * D_MODEL + j] = (y[r][j] - mean) * rstd * gamma[j] + beta[j];
    }
}

// ---------------------------------------------------------------------------
extern "C" void kernel_launch(void* const* d_in, const int* in_sizes, int n_in,
                              void* d_out, int out_size, void* d_ws, size_t ws_size,
                              hipStream_t stream) {
    (void)in_sizes; (void)n_in; (void)out_size; (void)ws_size;

    const float* x     = (const float*)d_in[0];
    const float* Wq    = (const float*)d_in[1];
    const float* bq    = (const float*)d_in[2];
    const float* Wk    = (const float*)d_in[3];
    const float* bk    = (const float*)d_in[4];
    const float* Wv    = (const float*)d_in[5];
    const float* bv    = (const float*)d_in[6];
    const float* Wo    = (const float*)d_in[7];
    const float* bo    = (const float*)d_in[8];
    const float* gamma = (const float*)d_in[9];
    const float* beta  = (const float*)d_in[10];
    float* out = (float*)d_out;

    // Workspace layout (bf16 scratch), ~44MB total.
    char* ws = (char*)d_ws;
    size_t off = 0;
    const size_t XBYTES = (size_t)ROWS * D_MODEL * sizeof(bf16_t);      // 8MB
    const size_t WBYTES = (size_t)D_MODEL * D_MODEL * sizeof(bf16_t);   // 512KB
    bf16_t* xb  = (bf16_t*)(ws + off); off += XBYTES;
    bf16_t* wqb = (bf16_t*)(ws + off); off += WBYTES;
    bf16_t* wkb = (bf16_t*)(ws + off); off += WBYTES;
    bf16_t* wvb = (bf16_t*)(ws + off); off += WBYTES;
    bf16_t* wob = (bf16_t*)(ws + off); off += WBYTES;
    bf16_t* qb  = (bf16_t*)(ws + off); off += XBYTES;   // [B,H,S,HD]
    bf16_t* kbuf= (bf16_t*)(ws + off); off += XBYTES;   // [B,H,S,HD]
    bf16_t* vt  = (bf16_t*)(ws + off); off += XBYTES;   // [B,H,HD,S]
    bf16_t* cb  = (bf16_t*)(ws + off); off += XBYTES;   // [B*S, D] context

    // 1) fp32 -> bf16 conversions
    prep_kernel<<<2048, 256, 0, stream>>>(x, Wq, Wk, Wv, Wo, xb, wqb, wkb, wvb, wob);

    // 2) QKV projections: 512 blocks, each owns one 16-row A-stripe in LDS
    qkv_gemm_kernel<<<ROWS / 16, 256, 0, stream>>>(xb, wqb, wkb, wvb, bq, bk, bv,
                                                   qb, kbuf, vt);

    // 3) attention: one WG per (b, h, 16-query stripe); 192KB dynamic LDS
    const int ATTN_LDS = 16 * SEQ * 4 + 16 * SEQ * 2;   // 196608 bytes
    hipFuncSetAttribute((const void*)attn_kernel,
                        hipFuncAttributeMaxDynamicSharedMemorySize, ATTN_LDS);
    attn_kernel<<<BATCH * NHEADS * (SEQ / 16), 128, ATTN_LDS, stream>>>(qb, kbuf, vt, cb);

    // 4) output projection + residual + LayerNorm
    out_ln_kernel<<<ROWS / 16, 256, 0, stream>>>(cb, wob, bo, x, gamma, beta, out);
}
